// ConvolutionalAttention3_6700148982327
// MI455X (gfx1250) — matrix operand order repaired and checked by reading, not compiled, not run on hardware
//
#include <hip/hip_runtime.h>

// Problem constants
#define BB 8
#define SS 2048
#define DD 512
#define MROWS (BB * SS)  // 16384

typedef __attribute__((ext_vector_type(16))) __bf16 bf16x16;
typedef __attribute__((ext_vector_type(8)))  float  f32x8;
typedef __attribute__((ext_vector_type(4))) unsigned int u32x4;
typedef __attribute__((ext_vector_type(8))) int          i32x8;
typedef __attribute__((ext_vector_type(4))) int          i32x4;

union Frag { bf16x16 v; unsigned u[8]; };

#if defined(__gfx1250__) && __has_builtin(__builtin_amdgcn_tensor_load_to_lds) && \
    __has_builtin(__builtin_amdgcn_s_wait_tensorcnt)
#define USE_TDM 1
#else
#define USE_TDM 0
#endif

__device__ __forceinline__ unsigned short f2bf(float f) {
  union { float f; unsigned u; } c; c.f = f;
  unsigned r = c.u + 0x7FFFu + ((c.u >> 16) & 1u);  // round-to-nearest-even
  return (unsigned short)(r >> 16);
}

__device__ __forceinline__ f32x8 zero8() {
  f32x8 z;
#pragma unroll
  for (int i = 0; i < 8; ++i) z[i] = 0.0f;
  return z;
}

__device__ __forceinline__ f32x8 wmma_bf16(const Frag& a, const Frag& b, f32x8 c) {
  return __builtin_amdgcn_wmma_f32_16x16x32_bf16(false, a.v, false, b.v, (short)0, c,
                                                 false, false);
}

// A fragment: 16(M) x 32(K) bf16 from row-major [m][k], row stride ld (elements).
// ISA layout: lanes 0-15 hold K pairs {0..7,16..23}, lanes 16-31 hold {8..15,24..31}.
__device__ __forceinline__ Frag load_a_frag(const unsigned short* base, int ld, int lane) {
  Frag f;
  const int hi = (lane >> 4) & 1;
  const unsigned short* row = base + (lane & 15) * ld;
#pragma unroll
  for (int i = 0; i < 8; ++i) {
    const int k = ((i < 4) ? 0 : 16) + hi * 8 + (i & 3) * 2;
    f.u[i] = *(const unsigned*)(row + k);
  }
  return f;
}

// B fragment: 32(K) x 16(N) bf16 stored N-major (Bt[n][k]), row stride ld.
// ISA layout: lanes 0-15 hold K {0..7,16..23}, lanes 16-31 hold {8..15,24..31}, N = lane&15.
__device__ __forceinline__ Frag load_b_frag(const unsigned short* base, int ld, int lane) {
  Frag f;
  const int hi = (lane >> 4) & 1;
  const unsigned short* row = base + (lane & 15) * ld;
#pragma unroll
  for (int i = 0; i < 8; ++i) {
    const int k = ((i < 4) ? 0 : 16) + hi * 8 + (i & 3) * 2;
    f.u[i] = *(const unsigned*)(row + k);
  }
  return f;
}

// ---- Tensor Data Mover: 2D bf16 tile load Global -> LDS (D# per ISA 08 §8.3/8.4) ----
// lds_off: wave-relative LDS byte address (low 32 bits of flat LDS pointer).
// Tile: tile_d0 (contiguous elems) x tile_d1 (rows); tensor extent tensor_d0 x tensor_d1,
// row stride stride_d0 (elements). data_size = 2 bytes (bf16).
__device__ __forceinline__ void tdm_load_2d_bf16(unsigned lds_off, const void* gptr,
                                                 unsigned tensor_d0, unsigned tensor_d1,
                                                 unsigned tile_d0, unsigned tile_d1,
                                                 unsigned stride_d0) {
#if USE_TDM
  unsigned long long ga = (unsigned long long)(uintptr_t)gptr;
  u32x4 g0;
  g0[0] = 1u;                                   // count=1, user mode
  g0[1] = lds_off;                              // lds_addr
  g0[2] = (unsigned)ga;                         // global_addr[31:0]
  g0[3] = ((unsigned)(ga >> 32) & 0x01FFFFFFu)  // global_addr[56:32]
          | (2u << 30);                         // type=2 ("image")
  i32x8 g1;
  g1[0] = (int)(1u << 16);                      // data_size=1 (2 bytes); mask/flags=0
  g1[1] = (int)((tensor_d0 & 0xFFFFu) << 16);   // tensor_dim0[15:0] at bits 63:48
  g1[2] = (int)((tensor_d0 >> 16) | ((tensor_d1 & 0xFFFFu) << 16)); // d0 hi | d1 lo
  g1[3] = (int)((tensor_d1 >> 16) | (tile_d0 << 16));               // d1 hi | tile_dim0
  g1[4] = (int)(tile_d1 & 0xFFFFu);             // tile_dim1; tile_dim2=0
  g1[5] = (int)stride_d0;                       // tensor_dim0_stride[31:0]
  g1[6] = 0;                                    // stride hi | tensor_dim1_stride lo
  g1[7] = 0;
  i32x4 z4 = {0, 0, 0, 0};
#if __clang_major__ >= 23
  i32x8 z8 = {0, 0, 0, 0, 0, 0, 0, 0};
  __builtin_amdgcn_tensor_load_to_lds(g0, g1, z4, z4, z8, 0);
#else
  __builtin_amdgcn_tensor_load_to_lds(g0, g1, z4, z4, 0);
#endif
#else
  (void)lds_off; (void)gptr; (void)tensor_d0; (void)tensor_d1;
  (void)tile_d0; (void)tile_d1; (void)stride_d0;
#endif
}

// ---------------- prep kernels ----------------

__global__ void cvt_f32_bf16(const float* __restrict__ X, unsigned short* __restrict__ Y,
                             int N) {
  int i = blockIdx.x * 256 + threadIdx.x;
  if (i < N) Y[i] = f2bf(X[i]);
}

// Wt[n][k] = W[k][n], 512x512, fp32 -> bf16
__global__ void wt_transpose(const float* __restrict__ W, unsigned short* __restrict__ Wt) {
  int idx = blockIdx.x * 256 + threadIdx.x;   // n*512 + k
  int n = idx >> 9, k = idx & 511;
  Wt[idx] = f2bf(W[k * 512 + n]);
}

// Vt[b][d][t] = v[b][t][d], fp32 -> bf16
__global__ void v_transpose(const float* __restrict__ V, unsigned short* __restrict__ Vt) {
  int idx = blockIdx.x * 256 + threadIdx.x;   // (b<<20) + (d<<11) + t
  int b = idx >> 20, rem = idx & 1048575;
  int d = rem >> 11, t = rem & 2047;
  Vt[idx] = f2bf(V[(b << 20) + t * 512 + d]);
}

// ---------------- WMMA GEMM: C[M,512] = A[M,512] @ W(=Bt N-major) + bias ----------------
// Block tile 64(M) x 128(N), 8 waves as 2x4, each wave 32x32 (2x2 fragments).

template <bool RELU, bool OUT_BF16>
__launch_bounds__(256)
__global__ void gemm_kernel(const unsigned short* __restrict__ A,
                            const unsigned short* __restrict__ Bt,
                            const float* __restrict__ bias,
                            void* __restrict__ Cout) {
  __shared__ __align__(16) unsigned short As[64][32];
  __shared__ __align__(16) unsigned short Bs[128][32];

  const int tid = threadIdx.x;
  const int lane = tid & 31, wave = tid >> 5;
  const int wm = wave >> 2, wn = wave & 3;
  const int mBase = blockIdx.x * 64;
  const int nBase = blockIdx.y * 128;

  f32x8 c[2][2];
  c[0][0] = zero8(); c[0][1] = zero8(); c[1][0] = zero8(); c[1][1] = zero8();

  const int ra = tid >> 2, ca = (tid & 3) * 8;           // A-tile chunk for this thread
  for (int k0 = 0; k0 < 512; k0 += 32) {
    // Stage A tile 64x32 (one uint4 / thread)
    *(uint4*)&As[ra][ca] = *(const uint4*)(A + (size_t)(mBase + ra) * 512 + k0 + ca);
    // Stage B tile 128x32 (two uint4 / thread)
#pragma unroll
    for (int rep = 0; rep < 2; ++rep) {
      int ch = tid + rep * 256;
      int r = ch >> 2, c8 = (ch & 3) * 8;
      *(uint4*)&Bs[r][c8] = *(const uint4*)(Bt + (size_t)(nBase + r) * 512 + k0 + c8);
    }
    // Prefetch next k-tile while this one computes (global_prefetch_b8)
    if (k0 + 32 < 512) {
      __builtin_prefetch(A + (size_t)(mBase + ra) * 512 + (k0 + 32) + ca, 0, 3);
      __builtin_prefetch(Bt + (size_t)(nBase + ra) * 512 + (k0 + 32) + ca, 0, 3);
      __builtin_prefetch(Bt + (size_t)(nBase + 64 + ra) * 512 + (k0 + 32) + ca, 0, 3);
    }
    __syncthreads();

    Frag a0 = load_a_frag(&As[wm * 32 + 0][0], 32, lane);
    Frag a1 = load_a_frag(&As[wm * 32 + 16][0], 32, lane);
    Frag b0 = load_b_frag(&Bs[wn * 32 + 0][0], 32, lane);
    Frag b1 = load_b_frag(&Bs[wn * 32 + 16][0], 32, lane);
    c[0][0] = wmma_bf16(a0, b0, c[0][0]);
    c[0][1] = wmma_bf16(a0, b1, c[0][1]);
    c[1][0] = wmma_bf16(a1, b0, c[1][0]);
    c[1][1] = wmma_bf16(a1, b1, c[1][1]);
    __syncthreads();
  }

  const int hi = lane >> 4, nl = lane & 15;
#pragma unroll
  for (int fi = 0; fi < 2; ++fi)
#pragma unroll
    for (int fj = 0; fj < 2; ++fj) {
      int col = nBase + wn * 32 + fj * 16 + nl;
      float bcol = bias[col];
#pragma unroll
      for (int i = 0; i < 8; ++i) {
        int row = mBase + wm * 32 + fi * 16 + hi * 8 + i;
        float r = c[fi][fj][i] + bcol;
        if (RELU) r = fmaxf(r, 0.0f);
        if (OUT_BF16)
          ((unsigned short*)Cout)[(size_t)row * 512 + col] = f2bf(r);
        else
          ((float*)Cout)[(size_t)row * 512 + col] = r;
      }
    }
}

// ---------------- LayerNorm + ReLU (fp32 in, bf16 out), one block per row ----------------

__launch_bounds__(256)
__global__ void ln_relu_kernel(const float* __restrict__ X, const float* __restrict__ g,
                               const float* __restrict__ beta,
                               unsigned short* __restrict__ Y) {
  __shared__ float red[256];
  const int row = blockIdx.x, tid = threadIdx.x;
  const float* x = X + (size_t)row * 512;
  float x0 = x[tid], x1 = x[tid + 256];

  red[tid] = x0 + x1;
  __syncthreads();
  for (int off = 128; off > 0; off >>= 1) {
    if (tid < off) red[tid] += red[tid + off];
    __syncthreads();
  }
  float mu = red[0] * (1.0f / 512.0f);
  __syncthreads();

  float d0 = x0 - mu, d1 = x1 - mu;
  red[tid] = d0 * d0 + d1 * d1;
  __syncthreads();
  for (int off = 128; off > 0; off >>= 1) {
    if (tid < off) red[tid] += red[tid + off];
    __syncthreads();
  }
  float var = red[0] * (1.0f / 512.0f);
  float rstd = rsqrtf(var + 1e-5f);

  float y0 = fmaxf(d0 * rstd * g[tid] + beta[tid], 0.0f);
  float y1 = fmaxf(d1 * rstd * g[tid + 256] + beta[tid + 256], 0.0f);
  Y[(size_t)row * 512 + tid] = f2bf(y0);
  Y[(size_t)row * 512 + tid + 256] = f2bf(y1);
}

// ---------------- Attention: scores, masked -softmax, attn@V ----------------
// One block per (batch b, 16 query rows). 8 waves; wave w owns key strip [w*256, w*256+256).
// Dynamic LDS: qT 16x512 bf16 | scF 16x2048 f32 | scB 16x2048 bf16 | red 256 f32 = 214016 B.

#define ATTN_SMEM (16 * 512 * 2 + 16 * 2048 * 4 + 16 * 2048 * 2 + 256 * 4)

__launch_bounds__(256)
__global__ void attn_kernel(const unsigned short* __restrict__ DQ,
                            const unsigned short* __restrict__ DK,
                            const unsigned short* __restrict__ Vt,
                            const unsigned char* __restrict__ mask,
                            float* __restrict__ outO,   // [B,S,512]
                            float* __restrict__ outA) { // [B,S,S]
  extern __shared__ char smem[];
  unsigned short* qT  = (unsigned short*)smem;                                  // 16x512
  float*          scF = (float*)(smem + 16 * 512 * 2);                          // 16x2048
  unsigned short* scB = (unsigned short*)(smem + 16 * 512 * 2 + 16 * 2048 * 4); // 16x2048
  float*          red = (float*)(smem + 16 * 512 * 2 + 16 * 2048 * 4 + 16 * 2048 * 2);

  const int s0 = blockIdx.x * 16;
  const int b  = blockIdx.y;
  const int tid = threadIdx.x, lane = tid & 31, wave = tid >> 5;
  const int hi = lane >> 4, nl = lane & 15;

  const unsigned short* DQb = DQ + (size_t)b * SS * DD;
  const unsigned short* DKb = DK + (size_t)b * SS * DD;
  const unsigned short* Vtb = Vt + (size_t)b * DD * SS;

  // Stage 16x512 query tile into LDS via the Tensor Data Mover (wave 0 issues the
  // DMA descriptor; TDM ignores EXEC and TENSORcnt is per-wave, so wave 0 waits
  // then the workgroup barrier publishes the tile). Fallback: vector loads.
#if USE_TDM
  if (wave == 0) {
    tdm_load_2d_bf16((unsigned)(uintptr_t)qT, DQb + (size_t)s0 * DD,
                     /*tensor_d0=*/DD, /*tensor_d1=*/SS,
                     /*tile_d0=*/DD, /*tile_d1=*/16, /*stride_d0=*/DD);
    __builtin_amdgcn_s_wait_tensorcnt(0);
  }
#else
#pragma unroll
  for (int rep = 0; rep < 4; ++rep) {
    int ch = tid + rep * 256;
    int r = ch >> 6, c8 = (ch & 63) * 8;
    *(uint4*)&qT[r * 512 + c8] = *(const uint4*)(DQb + (size_t)(s0 + r) * 512 + c8);
  }
#endif
  __syncthreads();

  // ---- scores = (Q @ K^T) / T ----
  Frag aq[16];
#pragma unroll
  for (int ks = 0; ks < 16; ++ks) aq[ks] = load_a_frag(qT + ks * 32, 512, lane);

  const float invT = 1.0f / 22.627416997969522f;
  for (int nt = 0; nt < 16; ++nt) {
    int tBase = wave * 256 + nt * 16;
    // Prefetch the next key strip while computing this one.
    if (nt + 1 < 16)
      __builtin_prefetch(DKb + (size_t)(tBase + 16 + (lane & 15)) * 512 + hi * 256, 0, 3);
    f32x8 c = zero8();
#pragma unroll
    for (int ks = 0; ks < 16; ++ks) {
      Frag bf = load_b_frag(DKb + (size_t)tBase * 512 + ks * 32, 512, lane);
      c = wmma_bf16(aq[ks], bf, c);
    }
#pragma unroll
    for (int i = 0; i < 8; ++i)
      scF[(hi * 8 + i) * 2048 + tBase + nl] = c[i] * invT;
  }
  __syncthreads();

  // ---- mask + negated softmax over t; 16 threads per row ----
  const int m = tid >> 4, j = tid & 15;
  const unsigned char* mrow = mask + ((size_t)b * SS + (s0 + m)) * SS;
  float* srow = scF + m * 2048;

  float lmax = -3.4e38f;
  for (int t = j; t < 2048; t += 16) {
    float v = mrow[t] ? -1e16f : srow[t];
    srow[t] = v;
    lmax = fmaxf(lmax, v);
  }
  red[tid] = lmax;
  __syncthreads();
  float rmax = red[m * 16];
  for (int q2 = 1; q2 < 16; ++q2) rmax = fmaxf(rmax, red[m * 16 + q2]);
  __syncthreads();

  float lsum = 0.0f;
  for (int t = j; t < 2048; t += 16) {
    float e = __expf(srow[t] - rmax);
    srow[t] = e;
    lsum += e;
  }
  red[tid] = lsum;
  __syncthreads();
  float rsum = 0.0f;
  for (int q2 = 0; q2 < 16; ++q2) rsum += red[m * 16 + q2];
  float rinv = 1.0f / rsum;

  float* arow = outA + ((size_t)b * SS + (s0 + m)) * SS;
  unsigned short* brow = scB + m * 2048;
  for (int t = j; t < 2048; t += 16) {
    float p = -(srow[t] * rinv);   // negated softmax (module quirk)
    arow[t] = p;
    brow[t] = f2bf(p);
  }
  __syncthreads();

  // ---- out = attn @ V ; wave w owns D columns [w*64, w*64+64) ----
  // A-fragment loaded once per k-step and reused for all 4 N-tiles.
  f32x8 acc[4];
#pragma unroll
  for (int nt = 0; nt < 4; ++nt) acc[nt] = zero8();
  const int n0w = wave * 64;
  for (int ks = 0; ks < 64; ++ks) {
    Frag af = load_a_frag(scB + ks * 32, 2048, lane);
#pragma unroll
    for (int nt = 0; nt < 4; ++nt) {
      Frag bf = load_b_frag(Vtb + (size_t)(n0w + nt * 16) * 2048 + ks * 32, 2048, lane);
      acc[nt] = wmma_bf16(af, bf, acc[nt]);
    }
  }
#pragma unroll
  for (int nt = 0; nt < 4; ++nt)
#pragma unroll
    for (int i = 0; i < 8; ++i) {
      int mr = hi * 8 + i;
      outO[((size_t)b * SS + s0 + mr) * 512 + n0w + nt * 16 + nl] = acc[nt][i];
    }
}

// ---------------- host side ----------------

extern "C" void kernel_launch(void* const* d_in, const int* in_sizes, int n_in,
                              void* d_out, int out_size, void* d_ws, size_t ws_size,
                              hipStream_t stream) {
  const float* q = (const float*)d_in[0];
  const float* k = (const float*)d_in[1];
  const float* v = (const float*)d_in[2];
  const unsigned char* mask = (const unsigned char*)d_in[3];
  const float* W[6]  = {(const float*)d_in[4], (const float*)d_in[5], (const float*)d_in[6],
                        (const float*)d_in[7], (const float*)d_in[8], (const float*)d_in[9]};
  const float* bia[6] = {(const float*)d_in[10], (const float*)d_in[11], (const float*)d_in[12],
                         (const float*)d_in[13], (const float*)d_in[14], (const float*)d_in[15]};
  const float* ln_b[2] = {(const float*)d_in[16], (const float*)d_in[17]};
  const float* ln_g[2] = {(const float*)d_in[18], (const float*)d_in[19]};

  char* ws = (char*)d_ws;
  size_t off = 0;
  auto alloc = [&](size_t bytes) {
    void* p = ws + off;
    off += (bytes + 255) & ~(size_t)255;
    return p;
  };
  unsigned short* Wt[6];
  for (int i = 0; i < 6; ++i) Wt[i] = (unsigned short*)alloc(512 * 512 * 2);
  unsigned short* Vt = (unsigned short*)alloc((size_t)BB * DD * SS * 2);
  unsigned short* DQ = (unsigned short*)alloc((size_t)MROWS * DD * 2);
  unsigned short* DK = (unsigned short*)alloc((size_t)MROWS * DD * 2);
  unsigned short* Xb = (unsigned short*)alloc((size_t)MROWS * DD * 2);
  unsigned short* H1 = (unsigned short*)alloc((size_t)MROWS * DD * 2);
  float*          H2 = (float*)alloc((size_t)MROWS * DD * 4);
  unsigned short* H3 = (unsigned short*)alloc((size_t)MROWS * DD * 2);

  float* outO = (float*)d_out;                        // [B,S,512]
  float* outA = (float*)d_out + (size_t)BB * SS * DD; // [B,S,S]

  // Prep: weights (transposed) and V (transposed), bf16
  for (int i = 0; i < 6; ++i)
    wt_transpose<<<(512 * 512) / 256, 256, 0, stream>>>(W[i], Wt[i]);
  v_transpose<<<(BB * DD * SS) / 256, 256, 0, stream>>>(v, Vt);

  const int NELT = MROWS * DD;
  const dim3 ggrid(MROWS / 64, 512 / 128);

  // Deep MLP for q (weights 0..2) and k (weights 3..5)
  for (int s = 0; s < 2; ++s) {
    const float* X = s ? k : q;
    unsigned short* Dx = s ? DK : DQ;
    unsigned short* const* Wtx = s ? (Wt + 3) : Wt;
    const float* const* bx = s ? (bia + 3) : bia;

    cvt_f32_bf16<<<NELT / 256, 256, 0, stream>>>(X, Xb, NELT);
    gemm_kernel<true, true><<<ggrid, 256, 0, stream>>>(Xb, Wtx[0], bx[0], H1);
    gemm_kernel<false, false><<<ggrid, 256, 0, stream>>>(H1, Wtx[1], bx[1], H2);
    ln_relu_kernel<<<MROWS, 256, 0, stream>>>(H2, ln_g[s], ln_b[s], H3);
    gemm_kernel<false, true><<<ggrid, 256, 0, stream>>>(H3, Wtx[2], bx[2], Dx);
  }

  // Attention (214 KB dynamic LDS — CDNA5 has 320 KB/WGP)
  (void)hipFuncSetAttribute((const void*)attn_kernel,
                            hipFuncAttributeMaxDynamicSharedMemorySize, ATTN_SMEM);
  attn_kernel<<<dim3(SS / 16, BB), 256, ATTN_SMEM, stream>>>(DQ, DK, Vt, mask, outO, outA);

  (void)in_sizes; (void)n_in; (void)out_size; (void)ws_size;
}
